// RWKV6_0_38560216384231
// MI455X (gfx1250) — compile-verified, matrix operand order checked by reading, not run on hardware
//
#include <hip/hip_runtime.h>
#include <hip/hip_bf16.h>
#include <cstdint>

// ---------------------------------------------------------------------------
// Types for CDNA5 WMMA
// ---------------------------------------------------------------------------
typedef __attribute__((ext_vector_type(16))) __bf16 v16bf;
typedef __attribute__((ext_vector_type(8)))  float  v8f;

#define LDT 40  // LDS row stride in bf16 elems (80B, 16B-aligned rows)

__device__ __forceinline__ unsigned f2bf(float f) {
  unsigned u = __builtin_bit_cast(unsigned, f);
  unsigned r = u + 0x7FFFu + ((u >> 16) & 1u);   // round-to-nearest-even
  return r >> 16;
}
__device__ __forceinline__ unsigned pk2bf(float lo, float hi) {
  return f2bf(lo) | (f2bf(hi) << 16);
}

struct B32x8 { uint4 lo, hi; };

// Load one 16-lane-row WMMA fragment from canonical LDS tile [row][k0..31].
// ISA 16-bit A/B layout: lanes 0-15 hold K=0..7 & 16..23, lanes 16-31 hold
// K=8..15 & 24..31 (kh = lane>>4 selects the 8-group).
__device__ __forceinline__ v16bf ldfrag(const unsigned short* rowp, int kh) {
  B32x8 u;
  u.lo = *reinterpret_cast<const uint4*>(rowp + kh * 8);
  u.hi = *reinterpret_cast<const uint4*>(rowp + 16 + kh * 8);
  return __builtin_bit_cast(v16bf, u);
}

__device__ __forceinline__ float act_apply(float v, int act) {
  switch (act) {
    case 1: return tanhf(v);
    case 2: { float r = v > 0.f ? v : 0.f; return r * r; }           // relu^2
    case 3: return v / (1.f + __expf(-v));                           // silu
    case 4: return 1.f / (1.f + __expf(-v));                         // sigmoid
    default: return v;
  }
}

// ---------------------------------------------------------------------------
// FAST GEMM (guard-free): C[M,N] = act( A[M,K] @ B[N,K]^T )
// Requires M%256==0, N%128==0, K%32==0, no bias, NK weight layout.
// Block tile 256x128, 8 waves as 4(m) x 2(n), each wave owns 64x64:
//   16 x v_wmma_f32_16x16x32_bf16 per K-chunk of 32.
// Double-buffered LDS + register prefetch of the next K-chunk.
// ---------------------------------------------------------------------------
__global__ __launch_bounds__(256)
void rwkv6_gemm_fast(const float* __restrict__ A, const float* __restrict__ Bm,
                     float* __restrict__ C, int N, int K, int lda, int ldb,
                     int act) {
  __shared__ __align__(16) unsigned short As[2][256 * LDT];  // 2 x 20 KB
  __shared__ __align__(16) unsigned short Bs[2][128 * LDT];  // 2 x 10 KB
  const int tid = threadIdx.x;
  const int m0 = blockIdx.y * 256;
  const int n0 = blockIdx.x * 128;
  const int wid = tid >> 5, lane = tid & 31;
  const int wm = wid & 3, wn = wid >> 2;   // wave tile: M[wm*64,+64) x N[wn*64,+64)
  const int row16 = lane & 15, kh = lane >> 4;

  // global load mapping: A row = tid (256 rows x 32 k), B row = tid>>1 half=(tid&1)
  const float* aRow = A + (size_t)(m0 + tid) * lda;
  const float* bRow = Bm + (size_t)(n0 + (tid >> 1)) * ldb + (tid & 1) * 16;

  unsigned ua[16], ub[8];  // packed bf16 prefetch (2 per uint)
  auto loadA = [&](int k0) {
#pragma unroll
    for (int q = 0; q < 8; q++) {
      float4 f = reinterpret_cast<const float4*>(aRow + k0)[q];
      ua[2 * q] = pk2bf(f.x, f.y); ua[2 * q + 1] = pk2bf(f.z, f.w);
    }
  };
  auto loadB = [&](int k0) {
#pragma unroll
    for (int q = 0; q < 4; q++) {
      float4 f = reinterpret_cast<const float4*>(bRow + k0)[q];
      ub[2 * q] = pk2bf(f.x, f.y); ub[2 * q + 1] = pk2bf(f.z, f.w);
    }
  };
  auto storeTiles = [&](int buf) {
    uint4* da = reinterpret_cast<uint4*>(As[buf] + tid * LDT);
#pragma unroll
    for (int q = 0; q < 4; q++)
      da[q] = make_uint4(ua[4 * q], ua[4 * q + 1], ua[4 * q + 2], ua[4 * q + 3]);
    uint4* db = reinterpret_cast<uint4*>(Bs[buf] + (tid >> 1) * LDT + (tid & 1) * 16);
#pragma unroll
    for (int q = 0; q < 2; q++)
      db[q] = make_uint4(ub[4 * q], ub[4 * q + 1], ub[4 * q + 2], ub[4 * q + 3]);
  };

  v8f acc[4][4];
#pragma unroll
  for (int i = 0; i < 4; i++)
#pragma unroll
    for (int j = 0; j < 4; j++)
#pragma unroll
      for (int e = 0; e < 8; e++) acc[i][j][e] = 0.f;

  loadA(0); loadB(0); storeTiles(0);
  __syncthreads();

  int buf = 0;
  for (int k0 = 0; k0 < K; k0 += 32) {
    const bool more = (k0 + 32) < K;
    if (more) { loadA(k0 + 32); loadB(k0 + 32); }  // overlap with WMMA below

    v16bf af[4], bf[4];
#pragma unroll
    for (int mt = 0; mt < 4; mt++)
      af[mt] = ldfrag(As[buf] + (wm * 64 + mt * 16 + row16) * LDT, kh);
#pragma unroll
    for (int nt = 0; nt < 4; nt++)
      bf[nt] = ldfrag(Bs[buf] + (wn * 64 + nt * 16 + row16) * LDT, kh);
#pragma unroll
    for (int mt = 0; mt < 4; mt++)
#pragma unroll
      for (int nt = 0; nt < 4; nt++)
        acc[mt][nt] = __builtin_amdgcn_wmma_f32_16x16x32_bf16(
            false, af[mt], false, bf[nt], (short)0, acc[mt][nt], false, false);

    if (more) storeTiles(buf ^ 1);
    __syncthreads();
    buf ^= 1;
  }

  // Epilogue: D layout -> VGPR e: lanes0-15 M=e, lanes16-31 M=e+8; N=lane&15
#pragma unroll
  for (int mt = 0; mt < 4; mt++) {
#pragma unroll
    for (int nt = 0; nt < 4; nt++) {
      int col = n0 + wn * 64 + nt * 16 + row16;
#pragma unroll
      for (int e = 0; e < 8; e++) {
        int row = m0 + wm * 64 + mt * 16 + e + 8 * kh;
        C[(size_t)row * N + col] = act_apply(acc[mt][nt][e], act);
      }
    }
  }
}

// ---------------------------------------------------------------------------
// Generic guarded GEMM (ragged / KN-layout / biased cases, off critical path)
//  blayout==0: B stored [N,K] row-major (A @ B^T); blayout==1: B stored [K,N]
// ---------------------------------------------------------------------------
__global__ __launch_bounds__(256)
void rwkv6_gemm(const float* __restrict__ A, const float* __restrict__ Bm,
                const float* __restrict__ bias, float* __restrict__ C,
                int M, int N, int K, int lda, int ldb, int blayout, int act) {
  __shared__ __align__(16) unsigned short As[128 * LDT];
  __shared__ __align__(16) unsigned short Bs[128 * LDT];
  const int tid = threadIdx.x;
  const int m0 = blockIdx.y * 128;
  const int n0 = blockIdx.x * 128;
  const int wid = tid >> 5, lane = tid & 31;
  const int wm = wid & 1, wn = wid >> 1;       // wave covers M[wm*64,+64) x N[wn*32,+32)
  const int row16 = lane & 15, kh = lane >> 4;

  v8f acc[4][2];
#pragma unroll
  for (int i = 0; i < 4; i++)
#pragma unroll
    for (int j = 0; j < 2; j++)
#pragma unroll
      for (int e = 0; e < 8; e++) acc[i][j][e] = 0.f;

  for (int k0 = 0; k0 < K; k0 += 32) {
    // ---- A tile: 128 rows x 32 k (K assumed multiple of 32) ----
    {
      int ar = tid >> 1, ac = (tid & 1) * 16;
      int grow = m0 + ar;
      unsigned short* dst = As + ar * LDT + ac;
      if (grow < M) {
        const float* src = A + (size_t)grow * lda + k0 + ac;
#pragma unroll
        for (int q = 0; q < 4; q++) {
          float4 f = reinterpret_cast<const float4*>(src)[q];
          dst[q * 4 + 0] = (unsigned short)f2bf(f.x);
          dst[q * 4 + 1] = (unsigned short)f2bf(f.y);
          dst[q * 4 + 2] = (unsigned short)f2bf(f.z);
          dst[q * 4 + 3] = (unsigned short)f2bf(f.w);
        }
      } else {
#pragma unroll
        for (int q = 0; q < 16; q++) dst[q] = 0;
      }
    }
    // ---- B tile -> canonical Bs[n][k] ----
    if (blayout == 0) {
      int br = tid >> 1, bc = (tid & 1) * 16;
      int gn = n0 + br;
      unsigned short* dst = Bs + br * LDT + bc;
      if (gn < N) {
        const float* src = Bm + (size_t)gn * ldb + k0 + bc;
#pragma unroll
        for (int q = 0; q < 4; q++) {
          float4 f = reinterpret_cast<const float4*>(src)[q];
          dst[q * 4 + 0] = (unsigned short)f2bf(f.x);
          dst[q * 4 + 1] = (unsigned short)f2bf(f.y);
          dst[q * 4 + 2] = (unsigned short)f2bf(f.z);
          dst[q * 4 + 3] = (unsigned short)f2bf(f.w);
        }
      } else {
#pragma unroll
        for (int q = 0; q < 16; q++) dst[q] = 0;
      }
    } else {
      int kk = tid >> 3, nb = (tid & 7) * 16;
      const float* src = Bm + (size_t)(k0 + kk) * ldb + n0 + nb;
      if (n0 + nb + 15 < N) {
#pragma unroll
        for (int q = 0; q < 4; q++) {
          float4 f = reinterpret_cast<const float4*>(src)[q];
          Bs[(nb + q * 4 + 0) * LDT + kk] = (unsigned short)f2bf(f.x);
          Bs[(nb + q * 4 + 1) * LDT + kk] = (unsigned short)f2bf(f.y);
          Bs[(nb + q * 4 + 2) * LDT + kk] = (unsigned short)f2bf(f.z);
          Bs[(nb + q * 4 + 3) * LDT + kk] = (unsigned short)f2bf(f.w);
        }
      } else {
#pragma unroll
        for (int q = 0; q < 16; q++) {
          int gn = n0 + nb + q;
          Bs[(nb + q) * LDT + kk] = (gn < N) ? (unsigned short)f2bf(src[q])
                                             : (unsigned short)0;
        }
      }
    }
    __syncthreads();

    v16bf afrag[4], bfrag[2];
#pragma unroll
    for (int mt = 0; mt < 4; mt++)
      afrag[mt] = ldfrag(As + (wm * 64 + mt * 16 + row16) * LDT, kh);
#pragma unroll
    for (int nt = 0; nt < 2; nt++)
      bfrag[nt] = ldfrag(Bs + (wn * 32 + nt * 16 + row16) * LDT, kh);

#pragma unroll
    for (int mt = 0; mt < 4; mt++)
#pragma unroll
      for (int nt = 0; nt < 2; nt++)
        acc[mt][nt] = __builtin_amdgcn_wmma_f32_16x16x32_bf16(
            false, afrag[mt], false, bfrag[nt], (short)0, acc[mt][nt],
            false, false);
    __syncthreads();
  }

#pragma unroll
  for (int mt = 0; mt < 4; mt++) {
#pragma unroll
    for (int nt = 0; nt < 2; nt++) {
      int col = n0 + wn * 32 + nt * 16 + row16;
#pragma unroll
      for (int e = 0; e < 8; e++) {
        int row = m0 + wm * 64 + mt * 16 + e + 8 * kh;
        if (row < M && col < N) {
          float v = acc[mt][nt][e];
          if (bias) v += bias[col];
          C[(size_t)row * N + col] = act_apply(v, act);
        }
      }
    }
  }
}

// ---------------------------------------------------------------------------
// WKV6 recurrence: one block per (b,h), 64 lanes, state row i in registers.
//  y_i = sum_j (S[i][j] + u_j k_j v_i) r_j ;  S[i][j] = S[i][j]*d_j + k_j v_i
// ---------------------------------------------------------------------------
__global__ __launch_bounds__(64)
void rwkv6_wkv(const float* __restrict__ r, const float* __restrict__ k,
               const float* __restrict__ v, const float* __restrict__ d,
               const float* __restrict__ u, const float* __restrict__ s0,
               float* __restrict__ y, int T, int H, int N) {
  int b = blockIdx.x / H, h = blockIdx.x % H;
  int i = threadIdx.x;
  __shared__ float rs[64], ks[64], ds[64], us[64];
  float S[64];
  const float* s0p = s0 + ((size_t)h * N + i) * N;
#pragma unroll
  for (int j = 0; j < 64; j++) S[j] = s0p[j];
  us[i] = u[h * N + i];
  __syncthreads();
  size_t stride = (size_t)H * N;
  size_t base = ((size_t)b * T * H + h) * N + i;
  for (int t = 0; t < T; t++) {
    size_t off = base + (size_t)t * stride;
    float vi = v[off];
    rs[i] = r[off]; ks[i] = k[off]; ds[i] = d[off];
    __syncthreads();
    float acc = 0.f;
#pragma unroll
    for (int j = 0; j < 64; j++) {
      float kv = ks[j] * vi;
      acc = fmaf(S[j] + us[j] * kv, rs[j], acc);
      S[j] = fmaf(S[j], ds[j], kv);
    }
    y[off] = acc;
    __syncthreads();
  }
}

// ---------------------------------------------------------------------------
// LayerNorm over last dim D (one block of 256 per row); in-place safe.
// ---------------------------------------------------------------------------
__global__ __launch_bounds__(256)
void rwkv6_layernorm(const float* __restrict__ x, const float* __restrict__ g,
                     const float* __restrict__ b, float* __restrict__ out,
                     int D, float eps) {
  int row = blockIdx.x, tid = threadIdx.x;
  const float* xr = x + (size_t)row * D;
  float sum = 0.f, sq = 0.f;
  for (int c = tid; c < D; c += 256) { float v = xr[c]; sum += v; sq += v * v; }
  __shared__ float s1[256], s2[256];
  s1[tid] = sum; s2[tid] = sq; __syncthreads();
  for (int st = 128; st > 0; st >>= 1) {
    if (tid < st) { s1[tid] += s1[tid + st]; s2[tid] += s2[tid + st]; }
    __syncthreads();
  }
  float m = s1[0] / D;
  float var = s2[0] / D - m * m;
  float rstd = rsqrtf(var + eps);
  for (int c = tid; c < D; c += 256)
    out[(size_t)row * D + c] = (xr[c] - m) * rstd * g[c] + b[c];
}

// ---------------------------------------------------------------------------
// GroupNorm: rows of C channels, G groups (16 groups of 64). In-place safe.
// ---------------------------------------------------------------------------
__global__ __launch_bounds__(256)
void rwkv6_groupnorm(const float* __restrict__ x, const float* __restrict__ g,
                     const float* __restrict__ b, float* __restrict__ out,
                     int C, int G, float eps) {
  int row = blockIdx.x, tid = threadIdx.x;
  int gs = C / G;                 // 64
  int grp = tid >> 4, s = tid & 15;   // 16 groups x 16 threads
  const float* xr = x + (size_t)row * C;
  float sum = 0.f, sq = 0.f;
  for (int u = s; u < gs; u += 16) { float v = xr[grp * gs + u]; sum += v; sq += v * v; }
  __shared__ float ssum[256], ssq[256];
  __shared__ float mu[16], rs[16];
  ssum[tid] = sum; ssq[tid] = sq; __syncthreads();
  if (s == 0) {
    float a = 0.f, c = 0.f;
    for (int q = 0; q < 16; q++) { a += ssum[grp * 16 + q]; c += ssq[grp * 16 + q]; }
    float m = a / gs;
    mu[grp] = m; rs[grp] = rsqrtf(c / gs - m * m + eps);
  }
  __syncthreads();
  for (int u = s; u < gs; u += 16) {
    int c0 = grp * gs + u;
    float v = (xr[c0] - mu[grp]) * rs[grp];
    out[(size_t)row * C + c0] = v * g[c0] + b[c0];
  }
}

// ---------------------------------------------------------------------------
// Elementwise / gather kernels
// ---------------------------------------------------------------------------
__global__ void rwkv6_embed(const int* __restrict__ idx,
                            const float* __restrict__ emb,
                            float* __restrict__ out, int C) {
  int row = blockIdx.x;
  const float* e = emb + (size_t)idx[row] * C;
  for (int c = threadIdx.x; c < C; c += blockDim.x)
    out[(size_t)row * C + c] = e[c];
}

__global__ void rwkv6_token_shift(const float* __restrict__ x,
                                  const float* __restrict__ st,
                                  float* __restrict__ xx, int T, int C, long n) {
  long i = (long)blockIdx.x * blockDim.x + threadIdx.x;
  if (i >= n) return;
  int c = (int)(i % C);
  long bt = i / C;
  int t = (int)(bt % T);
  float prev = (t == 0) ? st[c] : x[i - C];
  xx[i] = prev - x[i];
}

__global__ void rwkv6_xmix1(const float* __restrict__ x, const float* __restrict__ xx,
                            const float* __restrict__ mc, float* __restrict__ out,
                            int C, long n) {
  long i = (long)blockIdx.x * blockDim.x + threadIdx.x;
  if (i >= n) return;
  out[i] = x[i] + xx[i] * mc[i % C];
}

__global__ void rwkv6_xmix2(const float* __restrict__ x, const float* __restrict__ xx,
                            const float* __restrict__ mc, const float* __restrict__ md,
                            float* __restrict__ out, int C, long n) {
  long i = (long)blockIdx.x * blockDim.x + threadIdx.x;
  if (i >= n) return;
  out[i] = x[i] + xx[i] * (mc[i % C] + md[i]);
}

__global__ void rwkv6_tanh_ip(float* __restrict__ x, long n) {
  long i = (long)blockIdx.x * blockDim.x + threadIdx.x;
  if (i < n) x[i] = tanhf(x[i]);
}

__global__ void rwkv6_exp_neg_exp(const float* __restrict__ w, float* __restrict__ d, long n) {
  long i = (long)blockIdx.x * blockDim.x + threadIdx.x;
  if (i < n) d[i] = __expf(-__expf(w[i]));
}

__global__ void rwkv6_mul(const float* __restrict__ a, const float* __restrict__ b,
                          float* __restrict__ out, long n) {
  long i = (long)blockIdx.x * blockDim.x + threadIdx.x;
  if (i < n) out[i] = a[i] * b[i];
}

__global__ void rwkv6_add_ip(float* __restrict__ x, const float* __restrict__ d, long n) {
  long i = (long)blockIdx.x * blockDim.x + threadIdx.x;
  if (i < n) x[i] += d[i];
}

// ---------------------------------------------------------------------------
// Host orchestration
// ---------------------------------------------------------------------------
extern "C" void kernel_launch(void* const* d_in, const int* in_sizes, int n_in,
                              void* d_out, int out_size, void* d_ws, size_t ws_size,
                              hipStream_t stream) {
  (void)in_sizes; (void)out_size; (void)ws_size;
  constexpr int C = 1024, Aa = 1024, Fd = 3584, Vv = 50304;
  constexpr int H = 16, Nn = 64, L = 2, Bb = 2, T = 1024;
  constexpr int TMX = 32, TDX = 64, BT = Bb * T;
  if (n_in < 85) return;

  auto F = [&](int i) -> const float* { return (const float*)d_in[i]; };
  const int* idx = (const int*)d_in[0];
  const float* emb = F(1);

  // workspace bump allocator (fp32)
  float* Wp = (float*)d_ws;
  size_t off = 0;
  auto alloc = [&](size_t n) -> float* {
    float* p = Wp + off; off += (n + 63) & ~(size_t)63; return p;
  };
  float* x   = alloc((size_t)BT * C);
  float* xln = alloc((size_t)BT * C);
  float* xx  = alloc((size_t)BT * C);
  float* xw  = alloc((size_t)BT * C);
  float* xk  = alloc((size_t)BT * C);
  float* xv  = alloc((size_t)BT * C);
  float* xr  = alloc((size_t)BT * C);
  float* xg  = alloc((size_t)BT * C);
  float* rb  = alloc((size_t)BT * C);
  float* kb  = alloc((size_t)BT * C);
  float* vb  = alloc((size_t)BT * C);
  float* gb  = alloc((size_t)BT * C);
  float* wb  = alloc((size_t)BT * C);
  float* dec = alloc((size_t)BT * C);
  float* yb  = alloc((size_t)BT * C);
  float* mb  = alloc((size_t)BT * C);
  float* t5  = alloc((size_t)BT * 5 * TMX);
  float* dt  = alloc((size_t)BT * TDX);
  float* ffk = alloc((size_t)BT * Fd);
  float* h1  = alloc((size_t)H * 512);
  float* h2  = alloc((size_t)H * 4096);
  float* s0b = alloc((size_t)H * 4096);

  auto gemm = [&](const float* Am, const float* Bm, const float* bias, float* Cm,
                  int M, int N, int K, int lda, int ldb, int blayout, int act) {
    if (blayout == 0 && bias == nullptr && (M % 256) == 0 && (N % 128) == 0) {
      dim3 g((unsigned)(N / 128), (unsigned)(M / 256));
      rwkv6_gemm_fast<<<g, dim3(256), 0, stream>>>(Am, Bm, Cm, N, K, lda, ldb, act);
    } else {
      dim3 g((unsigned)((N + 127) / 128), (unsigned)((M + 127) / 128));
      rwkv6_gemm<<<g, dim3(256), 0, stream>>>(Am, Bm, bias, Cm, M, N, K, lda, ldb, blayout, act);
    }
  };
  const long nBTC = (long)BT * C;
  auto g1 = [&](long n) { return dim3((unsigned)((n + 255) / 256)); };

  // embedding gather
  rwkv6_embed<<<dim3(BT), dim3(256), 0, stream>>>(idx, emb, x, C);

  for (int l = 0; l < L; l++) {
    const int b0 = (l == 0) ? 2 : 43;
    const float *ln1g = F(b0 + 0), *ln1b = F(b0 + 1), *ln2g = F(b0 + 2), *ln2b = F(b0 + 3);
    const int a0 = b0 + 4;
    const float *tss = F(a0 + 0), *maa_x = F(a0 + 1), *maa_w = F(a0 + 2),
                *maa_k = F(a0 + 3), *maa_v = F(a0 + 4), *maa_r = F(a0 + 5),
                *maa_g = F(a0 + 6), *maa_w1 = F(a0 + 7), *maa_w2 = F(a0 + 8),
                *tdecay = F(a0 + 9), *dw1 = F(a0 + 10), *dw2 = F(a0 + 11),
                *faaaa = F(a0 + 12), *Wr = F(a0 + 13), *Wk = F(a0 + 14),
                *Wv = F(a0 + 15), *Wg = F(a0 + 16), *Wo = F(a0 + 17),
                *lnxg = F(a0 + 18), *lnxb = F(a0 + 19);
    const float *se_es = F(a0 + 20), *se_w1 = F(a0 + 21), *se_b1 = F(a0 + 22),
                *se_l1g = F(a0 + 23), *se_l1b = F(a0 + 24), *se_w2 = F(a0 + 25),
                *se_b2 = F(a0 + 26), *se_l2g = F(a0 + 27), *se_l2b = F(a0 + 28);
    const float *f_tss = F(a0 + 29), *f_mk = F(a0 + 30), *f_mr = F(a0 + 31),
                *f_Wk = F(a0 + 32), *f_Wr = F(a0 + 33), *f_Wv = F(a0 + 34);

    if (l == 0)  // ln0 applied once to the embedding
      rwkv6_layernorm<<<dim3(BT), dim3(256), 0, stream>>>(x, F(b0 + 39), F(b0 + 40), x, C, 1e-5f);

    // ===================== time-mix =====================
    rwkv6_layernorm<<<dim3(BT), dim3(256), 0, stream>>>(x, ln1g, ln1b, xln, C, 1e-5f);
    rwkv6_token_shift<<<g1(nBTC), dim3(256), 0, stream>>>(xln, tss, xx, T, C, nBTC);
    rwkv6_xmix1<<<g1(nBTC), dim3(256), 0, stream>>>(xln, xx, maa_x, mb, C, nBTC);
    // t5 = tanh((x + xx*maa_x) @ maa_w1)   (maa_w1 is [C,160] K-major)
    gemm(mb, maa_w1, nullptr, t5, BT, 5 * TMX, C, C, 5 * TMX, 1, 1);
    const float* mconst[5] = {maa_w, maa_k, maa_v, maa_r, maa_g};
    float* xdst[5] = {xw, xk, xv, xr, xg};
    for (int f = 0; f < 5; f++) {
      // m_f = t5[:,:,f,:] @ maa_w2[f]   (maa_w2[f] is [32,C] K-major)
      gemm(t5 + f * TMX, maa_w2 + (size_t)f * TMX * C, nullptr, mb,
           BT, C, TMX, 5 * TMX, C, 1, 0);
      rwkv6_xmix2<<<g1(nBTC), dim3(256), 0, stream>>>(xln, xx, mconst[f], mb, xdst[f], C, nBTC);
    }
    gemm(xr, Wr, nullptr, rb, BT, Aa, C, C, C, 0, 0);
    gemm(xk, Wk, nullptr, kb, BT, Aa, C, C, C, 0, 0);
    gemm(xv, Wv, nullptr, vb, BT, Aa, C, C, C, 0, 0);
    gemm(xg, Wg, nullptr, gb, BT, Aa, C, C, C, 0, 3);             // silu gate
    gemm(xw, dw1, nullptr, dt, BT, TDX, C, C, TDX, 1, 1);          // tanh(xw @ decay_w1)
    gemm(dt, dw2, tdecay, wb, BT, Aa, TDX, TDX, Aa, 1, 0);         // + time_decay
    rwkv6_exp_neg_exp<<<g1(nBTC), dim3(256), 0, stream>>>(wb, dec, nBTC);
    // state encoder: s0 = LN2(LN1_tanh(es@w1^T+b1)@w2^T+b2)
    gemm(se_es, se_w1, se_b1, h1, H, 512, 64, 64, 64, 0, 0);
    rwkv6_layernorm<<<dim3(H), dim3(256), 0, stream>>>(h1, se_l1g, se_l1b, h1, 512, 1e-5f);
    rwkv6_tanh_ip<<<g1((long)H * 512), dim3(256), 0, stream>>>(h1, (long)H * 512);
    gemm(h1, se_w2, se_b2, h2, H, 4096, 512, 512, 512, 0, 0);
    rwkv6_layernorm<<<dim3(H), dim3(256), 0, stream>>>(h2, se_l2g, se_l2b, s0b, 4096, 1e-5f);
    // sequential WKV scan
    rwkv6_wkv<<<dim3(Bb * H), dim3(Nn), 0, stream>>>(rb, kb, vb, dec, faaaa, s0b, yb, T, H, Nn);
    rwkv6_groupnorm<<<dim3(BT), dim3(256), 0, stream>>>(yb, lnxg, lnxb, yb, Aa, H, 6.4e-4f);
    rwkv6_mul<<<g1(nBTC), dim3(256), 0, stream>>>(yb, gb, mb, nBTC);
    gemm(mb, Wo, nullptr, xw, BT, C, Aa, Aa, Aa, 0, 0);
    rwkv6_add_ip<<<g1(nBTC), dim3(256), 0, stream>>>(x, xw, nBTC);

    // ===================== channel-mix =====================
    rwkv6_layernorm<<<dim3(BT), dim3(256), 0, stream>>>(x, ln2g, ln2b, xln, C, 1e-5f);
    rwkv6_token_shift<<<g1(nBTC), dim3(256), 0, stream>>>(xln, f_tss, xx, T, C, nBTC);
    rwkv6_xmix1<<<g1(nBTC), dim3(256), 0, stream>>>(xln, xx, f_mk, xk, C, nBTC);
    rwkv6_xmix1<<<g1(nBTC), dim3(256), 0, stream>>>(xln, xx, f_mr, xr, C, nBTC);
    gemm(xk, f_Wk, nullptr, ffk, BT, Fd, C, C, C, 0, 2);           // relu^2
    gemm(ffk, f_Wv, nullptr, mb, BT, C, Fd, Fd, Fd, 0, 0);
    gemm(xr, f_Wr, nullptr, xg, BT, C, C, C, C, 0, 4);             // sigmoid gate
    rwkv6_mul<<<g1(nBTC), dim3(256), 0, stream>>>(xg, mb, xw, nBTC);
    rwkv6_add_ip<<<g1(nBTC), dim3(256), 0, stream>>>(x, xw, nBTC);
  }

  // final LN + head projection (dominant GEMM, N = 50304)
  rwkv6_layernorm<<<dim3(BT), dim3(256), 0, stream>>>(x, F(82), F(83), xln, C, 1e-5f);
  gemm(xln, F(84), nullptr, (float*)d_out, BT, Vv, C, C, C, 0, 0);
}
